// CWTHead_8933531976384
// MI455X (gfx1250) — matrix operand order repaired and checked by reading, not compiled, not run on hardware
//
#include <hip/hip_runtime.h>
#include <hip/hip_bf16.h>

typedef __attribute__((ext_vector_type(16))) _Float16 v16h;
typedef __attribute__((ext_vector_type(8)))  float    v8f;
typedef __attribute__((ext_vector_type(4)))  float    f32x4;
typedef __attribute__((ext_vector_type(4)))  unsigned int u32x4;
typedef unsigned long long u64;

#define FREQS2    80           // 40 freqs * (re,im)
#define KTAPS     637
#define KPAD      640          // 20 * 32
#define KSTEPS    20
#define T_LEN     2000
#define PAD       318
#define XP_LEN    2636         // 2000 + 636
#define XP_STRIDE 2696         // >= 1536 + 1152 + 3, mult of 8
#define T_TILE    512          // 32 m-tiles per block = 4 per wave
#define TBLK      4            // ceil(2000 / 512)
#define XLEN      1152         // >= 512 + 636 + 3, mult of 8
#define WROW      648          // LDS row stride (halfs): 1296B, 16B aligned, bank-skewed

union AFrag { v16h v; u64   q[4]; };
union BFrag { v16h v; u32x4 u[2]; };

// cell = ROWS[c]*5 + COLS[c] per the reference _MAP
__device__ __constant__ int d_cell[32] = {
    1, 3, 5, 6, 7, 8, 9, 10, 11, 13, 14, 15, 16, 12, 18, 19,
    20, 21, 17, 22, 23, 24, 25, 26, 27, 28, 29, 30, 31, 32, 33, 34
};

// ---- prep: weights f32 -> f16, zero pad taps 637..639 ----
__global__ void prep_weights(const float* __restrict__ W, _Float16* __restrict__ wsW) {
    int idx = blockIdx.x * blockDim.x + threadIdx.x;
    if (idx >= FREQS2 * KPAD) return;
    int r = idx / KPAD, k = idx % KPAD;
    float v = (k < KTAPS) ? W[r * KTAPS + k] : 0.0f;
    wsW[idx] = (_Float16)v;
}

// ---- prep: reflect-padded signal f32 -> f16 ----
__global__ void prep_signal(const float* __restrict__ x, _Float16* __restrict__ wsX) {
    int idx = blockIdx.x * blockDim.x + threadIdx.x;
    if (idx >= 8 * 32 * XP_STRIDE) return;
    int i = idx % XP_STRIDE;
    int bc = idx / XP_STRIDE;
    float v = 0.0f;
    if (i < XP_LEN) {
        int j = i - PAD;
        j = (j < 0) ? -j : ((j >= T_LEN) ? (2 * T_LEN - 2 - j) : j);
        v = x[bc * T_LEN + j];
    }
    wsX[idx] = (_Float16)v;
}

// ---- prep: zero the 3 unmapped montage cells (cells 0,2,4) ----
__global__ void zero_gaps(float* __restrict__ out) {
    int idx = blockIdx.x * blockDim.x + threadIdx.x;
    if (idx >= 640 * 3 * T_LEN) return;      // 640 = B*2*F planes
    int t = idx % T_LEN;
    int rest = idx / T_LEN;
    int g = rest % 3;                        // gap 0,1,2 -> cell 0,2,4
    int pf = rest / 3;
    out[((size_t)pf * 35 + g * 2) * T_LEN + t] = 0.0f;
}

// ---- main: WMMA GEMM (4 m-tiles per wave, shared B, chained A overlap) ----
__global__ __launch_bounds__(256) void cwt_main(const _Float16* __restrict__ wsW,
                                                const _Float16* __restrict__ wsX,
                                                float* __restrict__ out) {
    __shared__ alignas(16) _Float16 Wl[16 * WROW];
    __shared__ alignas(16) _Float16 Xl[4][XLEN];

    int bid = blockIdx.x;
    int tb = bid % TBLK;
    int nt = (bid / TBLK) % 5;
    int c  = (bid / (TBLK * 5)) % 32;
    int b  = bid / (TBLK * 5 * 32);
    int t0 = tb * T_TILE;
    int mtiles = min(32, 125 - tb * 32);     // 32, except last t-block = 29

    // cooperative load: 16 filter rows (this block's n-tile), 640 taps each
    for (int u = threadIdx.x; u < 16 * (KPAD / 8); u += 256) {
        int r = u / (KPAD / 8), j = u % (KPAD / 8);
        *(u32x4*)&Wl[r * WROW + j * 8] =
            *(const u32x4*)&wsW[(nt * 16 + r) * KPAD + j * 8];
    }
    // cooperative load: signal window -> 4 shifted copies (1 gld + 4 lds-st / elem)
    const _Float16* xb = wsX + (size_t)(b * 32 + c) * XP_STRIDE + t0;
    for (int u = threadIdx.x; u < XLEN + 3; u += 256) {
        _Float16 v = xb[u];
        #pragma unroll
        for (int s = 0; s < 4; ++s) {
            int i = u - s;
            if (i >= 0 && i < XLEN) Xl[s][i] = v;
        }
    }
    __syncthreads();

    int wave = threadIdx.x >> 5;
    int lane = threadIdx.x & 31;
    int n    = lane & 15;        // N (filter within tile) / M-row lane
    int half = lane >> 4;        // K-half selector
    int sh   = lane & 3;         // A-run start parity

    int f2 = nt * 16 + n;
    int p  = f2 & 1;             // 0 = real->mag plane, 1 = imag->phase plane
    int f  = f2 >> 1;
    int cell = d_cell[c];

    int mtb = 4 * wave;          // four consecutive m-tiles per wave

    v8f acc0 = {}, acc1 = {}, acc2 = {}, acc3 = {};
    const _Float16* xrow = &Xl[sh][0];
    const _Float16* wrow = &Wl[n * WROW];

    #pragma unroll 2
    for (int ks = 0; ks < KSTEPS; ++ks) {
        // B (weights^T, 32x16 f16): depends only on (ks, lane) -> shared by 4 tiles
        BFrag bf;
        const u32x4* wq = (const u32x4*)(wrow + ks * 32 + 8 * half);
        bf.u[0] = wq[0];
        bf.u[1] = wq[2];
        // A tiles: tile i uses q-chunks {4i,4i+1,4i+4,4i+5}; union = 10 b64 loads
        int st = mtb * 16 + ks * 32 + n + 8 * half;
        const u64* xq = (const u64*)(xrow + (st - sh));
        u64 r0 = xq[0],  r1 = xq[1];
        u64 r2 = xq[4],  r3 = xq[5];
        u64 r4 = xq[8],  r5 = xq[9];
        u64 r6 = xq[12], r7 = xq[13];
        u64 r8 = xq[16], r9 = xq[17];
        AFrag a0; a0.q[0] = r0; a0.q[1] = r1; a0.q[2] = r2; a0.q[3] = r3;
        AFrag a1; a1.q[0] = r2; a1.q[1] = r3; a1.q[2] = r4; a1.q[3] = r5;
        AFrag a2; a2.q[0] = r4; a2.q[1] = r5; a2.q[2] = r6; a2.q[3] = r7;
        AFrag a3; a3.q[0] = r6; a3.q[1] = r7; a3.q[2] = r8; a3.q[3] = r9;

        acc0 = __builtin_amdgcn_wmma_f32_16x16x32_f16(
            false, a0.v, false, bf.v, (short)0, acc0, false, false);
        acc1 = __builtin_amdgcn_wmma_f32_16x16x32_f16(
            false, a1.v, false, bf.v, (short)0, acc1, false, false);
        acc2 = __builtin_amdgcn_wmma_f32_16x16x32_f16(
            false, a2.v, false, bf.v, (short)0, acc2, false, false);
        acc3 = __builtin_amdgcn_wmma_f32_16x16x32_f16(
            false, a3.v, false, bf.v, (short)0, acc3, false, false);
    }

    // pair re/im across adjacent lanes; even lane -> mag, odd lane -> phase
    #pragma unroll
    for (int which = 0; which < 4; ++which) {
        int mt = mtb + which;
        if (mt >= mtiles) continue;          // wave-uniform guard (stores only)
        v8f acc = (which == 0) ? acc0 : (which == 1) ? acc1
                : (which == 2) ? acc2 : acc3;
        float res[8];
        #pragma unroll
        for (int r = 0; r < 8; ++r) {
            float mine = acc[r];
            float oth  = __shfl_xor(mine, 1, 32);
            if (p == 0) res[r] = sqrtf(mine * mine + oth * oth);
            else        res[r] = atan2f(mine, oth);   // mine=im, oth=re
        }
        int tlane = t0 + mt * 16 + 8 * half;          // 8 consecutive t per lane
        size_t obase = ((size_t)((b * 2 + p) * 40 + f) * 35 + cell) * T_LEN + tlane;
        f32x4 lo = { res[0], res[1], res[2], res[3] };
        f32x4 hi = { res[4], res[5], res[6], res[7] };
        *(f32x4*)(out + obase)     = lo;
        *(f32x4*)(out + obase + 4) = hi;
    }
}

extern "C" void kernel_launch(void* const* d_in, const int* in_sizes, int n_in,
                              void* d_out, int out_size, void* d_ws, size_t ws_size,
                              hipStream_t stream) {
    const float* x = (const float*)d_in[0];
    const float* W = (const float*)d_in[1];
    float* out = (float*)d_out;

    _Float16* wsW = (_Float16*)d_ws;                                   // 80*640 halfs
    _Float16* wsX = (_Float16*)((char*)d_ws + (size_t)FREQS2 * KPAD * 2);

    int nW = FREQS2 * KPAD;
    prep_weights<<<(nW + 255) / 256, 256, 0, stream>>>(W, wsW);
    int nX = 8 * 32 * XP_STRIDE;
    prep_signal<<<(nX + 255) / 256, 256, 0, stream>>>(x, wsX);
    int nZ = 640 * 3 * T_LEN;
    zero_gaps<<<(nZ + 255) / 256, 256, 0, stream>>>(out);

    cwt_main<<<8 * 32 * 5 * TBLK, 256, 0, stream>>>(wsW, wsX, out);
}
